// SelfAttentionLayer_52664888984116
// MI455X (gfx1250) — matrix operand order repaired and checked by reading, not compiled, not run on hardware
//
#include <hip/hip_runtime.h>
#include <hip/hip_bf16.h>

namespace {

constexpr int Bc = 8, Tc = 24, Nc = 512, Dc = 128, Hc = 8, HIDc = 256;
constexpr long long ROWS = (long long)Bc * Tc * Nc;   // 98304
constexpr long long ELEM = ROWS * Dc;                 // 12,582,912

typedef __bf16 bf16_t;
typedef bf16_t v16bf __attribute__((ext_vector_type(16)));
typedef bf16_t v8bf  __attribute__((ext_vector_type(8)));
typedef float  v8f   __attribute__((ext_vector_type(8)));

// Epilogue modes
constexpr int EPI_F32  = 0;  // Cf = v
constexpr int EPI_BF   = 1;  // Cb = bf16(v)
constexpr int EPI_BOTH = 2;  // Cf = v, Cb = bf16(v)
constexpr int EPI_GLO0 = 3;  // Cf = xaux + attaux*v*scale   (x_glo init)
constexpr int EPI_GLO1 = 4;  // Cf += attaux*v*scale         (x_glo accum)

// ---------------------------------------------------------------------------
// WMMA GEMM: C[M,NCOL] = A[M,K] @ W[K,NCOL] (+bias) (+relu) (+epilogue)
// A : bf16 row-major [M,K]
// W : pre-transposed bf16 [NCOL,K] -> all fragment loads contiguous
// Each wave32 computes a 16x32 C tile (2 col tiles share the A fragment),
// K-loop in steps of 32 with v_wmma_f32_16x16x32_bf16.
// Batched via blockIdx.y: A += (batch%Amod)*strideA, W += batch*strideW,
// C += batch*strideC.
// ---------------------------------------------------------------------------
template<int K, int NCOL, bool BIAS, bool RELU, int EPI>
__global__ __launch_bounds__(256)
void wmma_gemm(const bf16_t* __restrict__ A, long long strideA, int Amod,
               const bf16_t* __restrict__ W, long long strideW,
               const float* __restrict__ bias,
               float* __restrict__ Cf, bf16_t* __restrict__ Cb,
               const float* __restrict__ attaux, const float* __restrict__ xaux,
               float scale, long long strideC, int M)
{
    const int batch = blockIdx.y;
    const bf16_t* Ab = A + (long long)(batch % Amod) * strideA;
    const bf16_t* Wb = W + (long long)batch * strideW;

    const int wave = threadIdx.x >> 5;
    const int lane = threadIdx.x & 31;
    const int half = lane >> 4;     // 0: lanes 0-15, 1: lanes 16-31
    const int r    = lane & 15;

    const int tilesN = NCOL / 32;
    const int tile = blockIdx.x * 8 + wave;
    if (tile >= (M / 16) * tilesN) return;
    const int tm = (tile / tilesN) * 16;
    const int tn = (tile % tilesN) * 32;
    const int c0 = tn + r, c1 = tn + 16 + r;

    const bf16_t* arow = Ab + (long long)(tm + r) * K;
    const bf16_t* wrow0 = Wb + (long long)c0 * K;
    const bf16_t* wrow1 = Wb + (long long)c1 * K;

    v8f acc0 = {}, acc1 = {};
    for (int k0 = 0; k0 < K; k0 += 32) {
        // A 16x32 fragment: elems 0..7 -> K=k0+16h+j, elems 8..15 -> K=k0+16h+8+j
        v8bf lo = *(const v8bf*)(arow + k0 + half * 16);
        v8bf hi = *(const v8bf*)(arow + k0 + 8 + half * 16);
        v16bf a;
        #pragma unroll
        for (int j = 0; j < 8; ++j) { a[j] = lo[j]; a[8 + j] = hi[j]; }

        // B 32x16 fragments: elem j -> K = k0 + 16h + j, col fixed per lane
        v16bf b0 = *(const v16bf*)(wrow0 + k0 + 16 * half);
        v16bf b1 = *(const v16bf*)(wrow1 + k0 + 16 * half);

        acc0 = __builtin_amdgcn_wmma_f32_16x16x32_bf16(
                   false, a, false, b0, (short)0, acc0, false, false);
        acc1 = __builtin_amdgcn_wmma_f32_16x16x32_bf16(
                   false, a, false, b1, (short)0, acc1, false, false);
    }

    float* Cfb = Cf ? Cf + (long long)batch * strideC : nullptr;
    bf16_t* Cbb = Cb ? Cb + (long long)batch * strideC : nullptr;
    const float bv0 = BIAS ? bias[c0] : 0.0f;
    const float bv1 = BIAS ? bias[c1] : 0.0f;
    #pragma unroll
    for (int i = 0; i < 8; ++i) {       // C: VGPR i -> row tm + i + 8*half
        const long long rowoff = (long long)(tm + i + 8 * half) * NCOL;
        const long long i0 = rowoff + c0, i1 = rowoff + c1;
        float v0 = acc0[i] + bv0;
        float v1 = acc1[i] + bv1;
        if (RELU) { v0 = v0 > 0.0f ? v0 : 0.0f; v1 = v1 > 0.0f ? v1 : 0.0f; }
        if (EPI == EPI_F32) {
            Cfb[i0] = v0; Cfb[i1] = v1;
        } else if (EPI == EPI_BF) {
            Cbb[i0] = (bf16_t)v0; Cbb[i1] = (bf16_t)v1;
        } else if (EPI == EPI_BOTH) {
            Cfb[i0] = v0; Cfb[i1] = v1;
            Cbb[i0] = (bf16_t)v0; Cbb[i1] = (bf16_t)v1;
        } else if (EPI == EPI_GLO0) {
            Cfb[i0] = xaux[i0] + attaux[i0] * v0 * scale;
            Cfb[i1] = xaux[i1] + attaux[i1] * v1 * scale;
        } else {  // EPI_GLO1
            Cfb[i0] = Cfb[i0] + attaux[i0] * v0 * scale;
            Cfb[i1] = Cfb[i1] + attaux[i1] * v1 * scale;
        }
    }
}

// f32 -> bf16 elementwise
__global__ __launch_bounds__(256)
void cvt_bf16(const float* __restrict__ src, bf16_t* __restrict__ dst, long long n)
{
    long long i = (long long)blockIdx.x * blockDim.x + threadIdx.x;
    if (i < n) dst[i] = (bf16_t)src[i];
}

// Transpose-convert weight: Wt[n*K + k] = (bf16)W[k*NCOL + n]
__global__ __launch_bounds__(256)
void wt_cvt(const float* __restrict__ W, bf16_t* __restrict__ Wt, int K, int NCOL)
{
    int idx = blockIdx.x * blockDim.x + threadIdx.x;
    if (idx >= K * NCOL) return;
    int n = idx / K, k = idx % K;
    Wt[idx] = (bf16_t)W[(long long)k * NCOL + n];
}

// Per-(b,t) transpose-convert of x: xT[bt][d][i] = (bf16)x[bt][i][d]
// 32x32 LDS tiles for coalescing. grid = (Nc/32, Dc/32, B*T), block = 256.
__global__ __launch_bounds__(256)
void xT_cvt(const float* __restrict__ x, bf16_t* __restrict__ xT)
{
    const int bt = blockIdx.z;
    const int i0 = blockIdx.x * 32;
    const int d0 = blockIdx.y * 32;
    __shared__ float tile[32][33];
    const int li = threadIdx.x & 31, lj = threadIdx.x >> 5;   // 32 x 8
    const float* src = x + (long long)bt * Nc * Dc;
    #pragma unroll
    for (int jj = lj; jj < 32; jj += 8)
        tile[jj][li] = src[(long long)(i0 + jj) * Dc + d0 + li];
    __syncthreads();
    bf16_t* dst = xT + (long long)bt * Dc * Nc;
    #pragma unroll
    for (int jj = lj; jj < 32; jj += 8)
        dst[(long long)(d0 + jj) * Nc + i0 + li] = (bf16_t)tile[li][jj];
}

// --------------------------------------------------------------------------
// L2-normalize q and k heads in-place inside the qkv buffer [ROWS, 384].
// --------------------------------------------------------------------------
__global__ __launch_bounds__(256)
void l2norm_qk(float* __restrict__ qkv)
{
    long long idx = (long long)blockIdx.x * blockDim.x + threadIdx.x;
    if (idx >= ROWS * Hc) return;
    int h = (int)(idx & 7);
    long long row = idx >> 3;
    float* q = qkv + row * 384 + h * 16;
    float* k = q + 128;
    float qv[16], kv[16], sq = 0.f, sk = 0.f;
    #pragma unroll
    for (int j = 0; j < 16; ++j) {
        qv[j] = q[j]; kv[j] = k[j];
        sq += qv[j] * qv[j]; sk += kv[j] * kv[j];
    }
    float rq = 1.0f / fmaxf(sqrtf(sq), 1e-12f);
    float rk = 1.0f / fmaxf(sqrtf(sk), 1e-12f);
    #pragma unroll
    for (int j = 0; j < 16; ++j) { q[j] = qv[j] * rq; k[j] = kv[j] * rk; }
}

// --------------------------------------------------------------------------
// Spatial: kvs[b,t,h,m,d] = sum_n kn[.,m]*v[.,d]; ksum[b,t,h,m] = sum_n kn[.,m]
// --------------------------------------------------------------------------
__global__ __launch_bounds__(256)
void spatial_kv(const float* __restrict__ qkv,
                float* __restrict__ kvs, float* __restrict__ ksum)
{
    const int bh = blockIdx.x;          // (b*T + t)*H + h
    const int h  = bh & 7;
    const long long bt = bh >> 3;
    const float* base = qkv + bt * (long long)Nc * 384;
    const int m = threadIdx.x >> 4, d = threadIdx.x & 15;

    __shared__ float sk[16][17];
    __shared__ float sv[16][17];
    float acc = 0.f, accs = 0.f;
    for (int n0 = 0; n0 < Nc; n0 += 16) {
        __syncthreads();
        sk[m][d] = base[(long long)(n0 + m) * 384 + 128 + h * 16 + d];
        sv[m][d] = base[(long long)(n0 + m) * 384 + 256 + h * 16 + d];
        __syncthreads();
        #pragma unroll
        for (int j = 0; j < 16; ++j) {
            acc  += sk[j][m] * sv[j][d];
            accs += sk[j][m];
        }
    }
    kvs[(long long)bh * 256 + m * 16 + d] = acc;
    if (d == 0) ksum[(long long)bh * 16 + m] = accs;
}

// --------------------------------------------------------------------------
// Temporal: kvt[b,n,h,m,d] = sum_t kn[.,m]*v[.,d]; ksumt[b,n,h,m]
// --------------------------------------------------------------------------
__global__ __launch_bounds__(256)
void temporal_kv(const float* __restrict__ qkv,
                 float* __restrict__ kvt, float* __restrict__ ksumt)
{
    const int id = blockIdx.x;          // (b*N + n)*H + h
    const int h = id & 7;
    const int bn = id >> 3;
    const int n = bn & (Nc - 1);
    const int b = bn >> 9;
    const int m = threadIdx.x >> 4, d = threadIdx.x & 15;

    __shared__ float sk[Tc][17];
    __shared__ float sv[Tc][17];
    for (int i = threadIdx.x; i < Tc * 16; i += 256) {
        int t = i >> 4, j = i & 15;
        long long row = ((long long)(b * Tc + t) * Nc + n);
        sk[t][j] = qkv[row * 384 + 128 + h * 16 + j];
        sv[t][j] = qkv[row * 384 + 256 + h * 16 + j];
    }
    __syncthreads();
    float acc = 0.f, accs = 0.f;
    #pragma unroll
    for (int t = 0; t < Tc; ++t) { acc += sk[t][m] * sv[t][d]; accs += sk[t][m]; }
    kvt[(long long)id * 256 + m * 16 + d] = acc;
    if (d == 0) ksumt[(long long)id * 16 + m] = accs;
}

// --------------------------------------------------------------------------
// Combine: out_s = (qn@kvs + N*v)/(qn.ksum + N), out_t analog with T.
// Writes attin as bf16 [ROWS,256] (consumed only by the out-proj GEMM).
// --------------------------------------------------------------------------
__global__ __launch_bounds__(256)
void attn_combine(const float* __restrict__ qkv,
                  const float* __restrict__ kvs, const float* __restrict__ ksums,
                  const float* __restrict__ kvt, const float* __restrict__ ksumt,
                  bf16_t* __restrict__ attin)
{
    long long idx = (long long)blockIdx.x * blockDim.x + threadIdx.x;
    if (idx >= ROWS * Hc) return;
    const int h = (int)(idx & 7);
    const long long row = idx >> 3;
    const int n = (int)(row & (Nc - 1));
    const long long bt = row >> 9;
    const int b = (int)(bt / Tc);

    const float* q = qkv + row * 384 + h * 16;
    const float* v = q + 256;
    const float* Ks = kvs   + (bt * 8 + h) * 256;
    const float* ss = ksums + (bt * 8 + h) * 16;
    const long long tid = ((long long)(b * Nc + n)) * 8 + h;
    const float* Kt = kvt   + tid * 256;
    const float* st = ksumt + tid * 16;

    float qv[16], vv[16];
    float dens = (float)Nc, dent = (float)Tc;
    #pragma unroll
    for (int m = 0; m < 16; ++m) {
        qv[m] = q[m]; vv[m] = v[m];
        dens += qv[m] * ss[m]; dent += qv[m] * st[m];
    }
    const float rds = 1.0f / dens, rdt = 1.0f / dent;
    bf16_t* os = attin + row * 256 + h * 16;
    bf16_t* ot = os + 128;
    #pragma unroll
    for (int d = 0; d < 16; ++d) {
        float ns = (float)Nc * vv[d], nt = (float)Tc * vv[d];
        #pragma unroll
        for (int m = 0; m < 16; ++m) {
            ns += qv[m] * Ks[m * 16 + d];
            nt += qv[m] * Kt[m * 16 + d];
        }
        os[d] = (bf16_t)(ns * rds);
        ot[d] = (bf16_t)(nt * rdt);
    }
}

// out = LayerNorm(a + b) * g + be ; optional bf16 mirror for GEMM consumer
template<bool WBF>
__global__ __launch_bounds__(128)
void add_layernorm(const float* __restrict__ a, const float* __restrict__ b,
                   const float* __restrict__ g, const float* __restrict__ be,
                   float* __restrict__ out, bf16_t* __restrict__ outb)
{
    const long long row = blockIdx.x;
    const int j = threadIdx.x;
    __shared__ float red[128];
    float v = a[row * Dc + j] + b[row * Dc + j];
    red[j] = v; __syncthreads();
    for (int s = 64; s > 0; s >>= 1) { if (j < s) red[j] += red[j + s]; __syncthreads(); }
    const float mu = red[0] / (float)Dc;
    __syncthreads();
    const float dv = v - mu;
    red[j] = dv * dv; __syncthreads();
    for (int s = 64; s > 0; s >>= 1) { if (j < s) red[j] += red[j + s]; __syncthreads(); }
    const float inv = rsqrtf(red[0] / (float)Dc + 1e-5f);
    const float o = dv * inv * g[j] + be[j];
    out[row * Dc + j] = o;
    if (WBF) outb[row * Dc + j] = (bf16_t)o;
}

} // namespace

extern "C" void kernel_launch(void* const* d_in, const int* in_sizes, int n_in,
                              void* d_out, int out_size, void* d_ws, size_t ws_size,
                              hipStream_t stream)
{
    (void)in_sizes; (void)n_in; (void)out_size; (void)ws_size;
    const float* x      = (const float*)d_in[0];
    const float* graph  = (const float*)d_in[1];
    const float* qkv_w[2] = { (const float*)d_in[2], (const float*)d_in[7] };
    const float* out_w[2] = { (const float*)d_in[3], (const float*)d_in[8] };
    const float* out_b[2] = { (const float*)d_in[4], (const float*)d_in[9] };
    const float* pw_w[2]  = { (const float*)d_in[5], (const float*)d_in[10] };
    const float* pw_b[2]  = { (const float*)d_in[6], (const float*)d_in[11] };
    const float* fc1_w = (const float*)d_in[12];
    const float* fc1_b = (const float*)d_in[13];
    const float* fc2_w = (const float*)d_in[14];
    const float* fc2_b = (const float*)d_in[15];
    const float* ln1_g = (const float*)d_in[16];
    const float* ln1_b = (const float*)d_in[17];
    const float* ln2_g = (const float*)d_in[18];
    const float* ln2_b = (const float*)d_in[19];
    float* out = (float*)d_out;

    // ---- workspace layout (64B-aligned bump allocator over floats) ----
    float* ws = (float*)d_ws;
    size_t off = 0;
    auto allocF = [&](size_t nf) {
        off = (off + 15) & ~(size_t)15; float* p = ws + off; off += nf; return p;
    };
    auto allocB = [&](size_t nb) {
        off = (off + 15) & ~(size_t)15; bf16_t* p = (bf16_t*)(ws + off);
        off += (nb + 1) / 2; return p;
    };

    // f32 buffers
    float* qkvb  = allocF(ROWS * 384);
    float* att0  = allocF(ELEM);
    float* att1  = allocF(ELEM);
    float* xglo  = allocF(ELEM);
    float* kvs   = allocF((size_t)Bc * Tc * Hc * 256);
    float* ksums = allocF((size_t)Bc * Tc * Hc * 16);
    float* kvt   = allocF((size_t)Bc * Nc * Hc * 256);
    float* ksumt = allocF((size_t)Bc * Nc * Hc * 16);
    float* hbuf  = allocF(ELEM);
    float* mlp2  = allocF(ELEM);
    // bf16 buffers
    bf16_t* graph_bf = allocB((size_t)Tc * Nc * Nc);
    bf16_t* x_bf     = allocB(ELEM);       // row-major [ROWS,128]
    bf16_t* xT_bf    = allocB(ELEM);       // per-(b,t) transposed [128,512]
    bf16_t* x1_bf    = allocB(ELEM);
    bf16_t* attin_bf = allocB(ROWS * 256);
    bf16_t* att0_bf  = allocB(ELEM);
    bf16_t* h_bf     = allocB(ELEM);
    bf16_t* mlp1_bf  = allocB(ROWS * 256);
    // transposed bf16 weights [NCOL,K]
    bf16_t* qkv_wT[2] = { allocB(128 * 384), allocB(128 * 384) };
    bf16_t* out_wT[2] = { allocB(256 * 128), allocB(256 * 128) };
    bf16_t* pw_wT[2]  = { allocB(128 * 128), allocB(128 * 128) };
    bf16_t* fc1_wT    = allocB(128 * 256);
    bf16_t* fc2_wT    = allocB(256 * 128);

    const dim3 blk256(256);
    auto cdiv = [](long long a, long long b) { return (int)((a + b - 1) / b); };

    // ---- 0) operand conversions ----
    wt_cvt<<<cdiv(128 * 384, 256), blk256, 0, stream>>>(qkv_w[0], qkv_wT[0], 128, 384);
    wt_cvt<<<cdiv(128 * 384, 256), blk256, 0, stream>>>(qkv_w[1], qkv_wT[1], 128, 384);
    wt_cvt<<<cdiv(256 * 128, 256), blk256, 0, stream>>>(out_w[0], out_wT[0], 256, 128);
    wt_cvt<<<cdiv(256 * 128, 256), blk256, 0, stream>>>(out_w[1], out_wT[1], 256, 128);
    wt_cvt<<<cdiv(128 * 128, 256), blk256, 0, stream>>>(pw_w[0], pw_wT[0], 128, 128);
    wt_cvt<<<cdiv(128 * 128, 256), blk256, 0, stream>>>(pw_w[1], pw_wT[1], 128, 128);
    wt_cvt<<<cdiv(128 * 256, 256), blk256, 0, stream>>>(fc1_w, fc1_wT, 128, 256);
    wt_cvt<<<cdiv(256 * 128, 256), blk256, 0, stream>>>(fc2_w, fc2_wT, 256, 128);
    cvt_bf16<<<cdiv(ELEM, 256), blk256, 0, stream>>>(x, x_bf, ELEM);
    cvt_bf16<<<cdiv((long long)Tc * Nc * Nc, 256), blk256, 0, stream>>>(
        graph, graph_bf, (long long)Tc * Nc * Nc);
    xT_cvt<<<dim3(Nc / 32, Dc / 32, Bc * Tc), blk256, 0, stream>>>(x, xT_bf);

    // GEMM grids: tiles = (M/16)*(NCOL/32), 8 waves per block
    const int tilesRows = (int)(ROWS / 16);                 // 6144
    const int gQKV  = tilesRows * (384 / 32) / 8;           // 9216
    const int gOUT  = tilesRows * (128 / 32) / 8;           // 3072
    const int gPW   = gOUT;
    const int gFC1  = tilesRows * (256 / 32) / 8;           // 6144
    const int gFC2  = gOUT;
    const int blkRowsH = cdiv(ROWS * Hc, 256);

    // ---- 1) graph propagation: x1_bf[b,t] = graph[t] @ x[b,t] (batched) ----
    wmma_gemm<Nc, Dc, false, false, EPI_BF>
        <<<dim3((512 / 16) * (128 / 32) / 8, Bc * Tc), blk256, 0, stream>>>(
        graph_bf, (long long)Nc * Nc, Tc,
        xT_bf, (long long)Dc * Nc, nullptr,
        nullptr, x1_bf, nullptr, nullptr, 0.0f, (long long)Nc * Dc, Nc);

    const bf16_t* attn_src[2] = { x_bf, x1_bf };
    const bf16_t* pw_src[2]   = { x_bf, att0_bf };
    float* att_dst[2] = { att0, att1 };

    for (int L = 0; L < 2; ++L) {
        // qkv = attn_src @ qkv_w  -> f32 (feeds l2norm/attention VALU kernels)
        wmma_gemm<Dc, 384, false, false, EPI_F32>
            <<<dim3(gQKV, 1), blk256, 0, stream>>>(
            attn_src[L], 0, 1, qkv_wT[L], 0, nullptr,
            qkvb, nullptr, nullptr, nullptr, 0.0f, 0, (int)ROWS);
        l2norm_qk<<<blkRowsH, blk256, 0, stream>>>(qkvb);
        spatial_kv<<<Bc * Tc * Hc, blk256, 0, stream>>>(qkvb, kvs, ksums);
        temporal_kv<<<Bc * Nc * Hc, blk256, 0, stream>>>(qkvb, kvt, ksumt);
        attn_combine<<<blkRowsH, blk256, 0, stream>>>(qkvb, kvs, ksums, kvt, ksumt, attin_bf);
        // att = attin @ out_w + out_b; L0 also emits bf16 mirror for L1 pw GEMM
        if (L == 0)
            wmma_gemm<256, Dc, true, false, EPI_BOTH>
                <<<dim3(gOUT, 1), blk256, 0, stream>>>(
                attin_bf, 0, 1, out_wT[0], 0, out_b[0],
                att0, att0_bf, nullptr, nullptr, 0.0f, 0, (int)ROWS);
        else
            wmma_gemm<256, Dc, true, false, EPI_F32>
                <<<dim3(gOUT, 1), blk256, 0, stream>>>(
                attin_bf, 0, 1, out_wT[1], 0, out_b[1],
                att1, nullptr, nullptr, nullptr, 0.0f, 0, (int)ROWS);
        // pw GEMM with fused x_glo epilogue:
        //   L0: xglo = x + att0 * pw * 1.0 ; L1: xglo += att1 * pw * 0.01
        if (L == 0)
            wmma_gemm<Dc, Dc, true, false, EPI_GLO0>
                <<<dim3(gPW, 1), blk256, 0, stream>>>(
                pw_src[0], 0, 1, pw_wT[0], 0, pw_b[0],
                xglo, nullptr, att_dst[0], x, 1.0f, 0, (int)ROWS);
        else
            wmma_gemm<Dc, Dc, true, false, EPI_GLO1>
                <<<dim3(gPW, 1), blk256, 0, stream>>>(
                pw_src[1], 0, 1, pw_wT[1], 0, pw_b[1],
                xglo, nullptr, att_dst[1], nullptr, 0.01f, 0, (int)ROWS);
    }

    // h = LN(x + x_glo)  (f32 + bf16 mirror)
    add_layernorm<true><<<(int)ROWS, 128, 0, stream>>>(x, xglo, ln1_g, ln1_b, hbuf, h_bf);
    // mlp1 = relu(h @ fc1_w + fc1_b) -> bf16 directly
    wmma_gemm<Dc, HIDc, true, true, EPI_BF>
        <<<dim3(gFC1, 1), blk256, 0, stream>>>(
        h_bf, 0, 1, fc1_wT, 0, fc1_b,
        nullptr, mlp1_bf, nullptr, nullptr, 0.0f, 0, (int)ROWS);
    // mlp2 = mlp1 @ fc2_w + fc2_b -> f32
    wmma_gemm<HIDc, Dc, true, false, EPI_F32>
        <<<dim3(gFC2, 1), blk256, 0, stream>>>(
        mlp1_bf, 0, 1, fc2_wT, 0, fc2_b,
        mlp2, nullptr, nullptr, nullptr, 0.0f, 0, (int)ROWS);
    // out = LN(h + mlp2)
    add_layernorm<false><<<(int)ROWS, 128, 0, stream>>>(hbuf, mlp2, ln2_g, ln2_b, out, nullptr);
}
